// Conv4DBlock_85624468013641
// MI455X (gfx1250) — matrix operand order repaired and checked by reading, not compiled, not run on hardware
//
#include <hip/hip_runtime.h>

typedef __attribute__((ext_vector_type(16))) _Float16 v16h;
typedef __attribute__((ext_vector_type(8)))  _Float16 v8h;
typedef __attribute__((ext_vector_type(8)))  float    v8f;

#define CI    16
#define LI    32
#define SI    24
#define CO    32
#define LO    30
#define SO    22
#define TAPS  81

#define XPATCH  (CI*3*3*SI*SI)            // 82944 f16 elements
#define WELEMS  (CO*CI*TAPS)              // 41472 f16 elements
#define SMEM_X  (XPATCH*2)                // 165888 B
#define SMEM_W  (WELEMS*2)                // 82944 B
#define SMEM_T  (84*4)                    // tap-offset table (81 used)
#define SMEM_TOTAL (SMEM_X + SMEM_W + SMEM_T)   // 249168 B < 320 KB/WGP

#define WMMA16(a, b, c) \
    __builtin_amdgcn_wmma_f32_16x16x32_f16(false, (a), false, (b), (short)0, (c), false, false)

#define CAT16(lo, hi) __builtin_shufflevector((lo), (hi), \
    0,1,2,3,4,5,6,7,8,9,10,11,12,13,14,15)

__global__ __launch_bounds__(256, 1)
void conv4d_wmma_f16(const float* __restrict__ xg,
                     const float* __restrict__ wg,
                     const float* __restrict__ bg,
                     float* __restrict__ outg)
{
    extern __shared__ char smem[];
    _Float16* sX = (_Float16*)smem;                       // [l'][d'][h][w][c]
    _Float16* sW = (_Float16*)(smem + SMEM_X);            // [tap][o][c]
    int*      sT = (int*)(smem + SMEM_X + SMEM_W);        // tapOff[tap] (f16 elems)

    const int blk = blockIdx.x;                           // 8*30*22 blocks
    const int dd  = blk % SO;
    const int l   = (blk / SO) % LO;
    const int n   = blk / (SO * LO);

    const int tid = threadIdx.x;

    // ---- Phase 1: stage input patch f32 -> f16 into LDS ----
    for (int e = tid; e < XPATCH; e += 256) {
        const int wq  = e % SI;
        const int hq  = (e / SI) % SI;
        const int dpr = (e / (SI*SI)) % 3;
        const int lpr = (e / (SI*SI*3)) % 3;
        const int c   = e / (SI*SI*9);
        const size_t gi = ((((size_t)(n*CI + c)*LI + (l + lpr))*SI + (dd + dpr))*SI + hq)*SI + wq;
        const int    si = (((lpr*3 + dpr)*SI + hq)*SI + wq)*CI + c;
        sX[si] = (_Float16)xg[gi];
    }
    // ---- Phase 2: stage weights f32 -> f16, layout [tap][o][c]; tap offset table ----
    for (int e = tid; e < WELEMS; e += 256) {
        const int o   = e / (CI*TAPS);
        const int i   = (e / TAPS) % CI;
        const int tap = e % TAPS;
        sW[(tap*CO + o)*CI + i] = (_Float16)wg[e];
    }
    if (tid < TAPS) {
        int r = tid;
        const int lt = r / 27;  r -= lt*27;
        const int dt = r / 9;   r -= dt*9;
        const int ht = r / 3;
        const int wt = r - ht*3;
        sT[tid] = ((lt*3 + dt)*SI*SI + ht*SI + wt) * CI;  // B addr is base(oh,ow)+this
    }
    __syncthreads();

    const int wave = tid >> 5;
    const int lane = tid & 31;
    const int col  = lane & 15;       // N column of the 16x16 tile
    const int hiH  = lane >> 4;       // 0 => K 0..7,16..23 lane group, 1 => K 8..15,24..31
    const int cb   = hiH ? 8 : 0;     // A-fragment c_in sub-range for this lane

    // ---- Each wave owns 4 position tiles (N=16) x 2 channel tiles ----
    int  baseoff[4];                  // (oh*24+ow)*16 per tile, this lane's column
    int  ohA[4], owA[4];
    bool stOK[4];
    #pragma unroll
    for (int j = 0; j < 4; ++j) {
        const int torig = 4*wave + j;               // 0..31; 31 tiles exist
        const int t     = torig < 31 ? torig : 30;  // clamp (compute-only duplicate)
        const int pos   = t*16 + col;
        const int cpos  = pos < SO*SO ? pos : SO*SO - 1;
        ohA[j] = cpos / SO;  owA[j] = cpos % SO;
        baseoff[j] = (ohA[j]*SI + owA[j]) * CI;
        stOK[j] = (torig < 31) && (pos < SO*SO);
    }

    v8f accLo[4] = {{},{},{},{}};
    v8f accHi[4] = {{},{},{},{}};

    // ---- Hot loop: taps 0..79, two taps (K=32) per step, branch-free ----
    for (int kt = 0; kt < 40; ++kt) {
        const int tapA = 2*kt;
        const int toff  = sT[tapA];                 // B tap offsets (table-order K)
        const int toff1 = sT[tapA + 1];

        const _Float16* aw = sW + tapA*(CO*CI) + cb;
        const v8h a0lo = *(const v8h*)(aw + col*CI);
        const v8h a0hi = *(const v8h*)(aw + CO*CI + col*CI);
        const v8h a1lo = *(const v8h*)(aw + (16 + col)*CI);
        const v8h a1hi = *(const v8h*)(aw + CO*CI + (16 + col)*CI);
        const v16h af0 = CAT16(a0lo, a0hi);
        const v16h af1 = CAT16(a1lo, a1hi);

        #pragma unroll
        for (int j = 0; j < 4; ++j) {
            const _Float16* bq = sX + baseoff[j] + cb;
            const v16h bf = CAT16(*(const v8h*)(bq + toff), *(const v8h*)(bq + toff1));
            accLo[j] = WMMA16(af0, bf, accLo[j]);
            accHi[j] = WMMA16(af1, bf, accHi[j]);
        }
    }

    // ---- Tail: tap 80 alone (upper K half zero-padded) ----
    {
        const v8h z8 = {};
        const int toff80 = sT[80];
        const _Float16* aw = sW + 80*(CO*CI) + cb;
        const v16h af0 = CAT16(*(const v8h*)(aw + col*CI), z8);
        const v16h af1 = CAT16(*(const v8h*)(aw + (16 + col)*CI), z8);
        #pragma unroll
        for (int j = 0; j < 4; ++j) {
            v8h b0 = *(const v8h*)(sX + baseoff[j] + toff80 + cb), b1 = z8;
            const v16h bf = CAT16(b0, b1);
            accLo[j] = WMMA16(af0, bf, accLo[j]);
            accHi[j] = WMMA16(af1, bf, accHi[j]);
        }
    }

    // ---- Epilogue: D layout lane => N=col, VGPR r => M=r+8*hiH; bias*3 ----
    const size_t chanStride = (size_t)LO*SO*SO*SO;
    #pragma unroll
    for (int j = 0; j < 4; ++j) {
        if (stOK[j]) {
            const size_t base = ((((size_t)n*CO)*LO + l)*SO + dd)*SO*SO
                              + (size_t)ohA[j]*SO + owA[j];
            #pragma unroll
            for (int r2 = 0; r2 < 8; ++r2) {
                const int m = r2 + 8*hiH;
                outg[base + (size_t)m*chanStride]        = accLo[j][r2] + 3.0f*bg[m];
                outg[base + (size_t)(16 + m)*chanStride] = accHi[j][r2] + 3.0f*bg[16 + m];
            }
        }
    }
}

extern "C" void kernel_launch(void* const* d_in, const int* in_sizes, int n_in,
                              void* d_out, int out_size, void* d_ws, size_t ws_size,
                              hipStream_t stream) {
    (void)in_sizes; (void)n_in; (void)out_size; (void)d_ws; (void)ws_size;
    const float* x = (const float*)d_in[0];
    const float* w = (const float*)d_in[1];
    const float* b = (const float*)d_in[2];
    float* out = (float*)d_out;

    // 249 KB dynamic LDS > default 64 KB cap (host-side attr: graph-capture safe).
    hipFuncSetAttribute((const void*)conv4d_wmma_f16,
                        hipFuncAttributeMaxDynamicSharedMemorySize, SMEM_TOTAL);

    const int nblocks = 8 * LO * SO;   // one block per (n, l, d) output plane
    conv4d_wmma_f16<<<dim3(nblocks), dim3(256), SMEM_TOTAL, stream>>>(x, w, b, out);
}
